// MultiheadLinearAttention_58059367907818
// MI455X (gfx1250) — matrix operand order repaired and checked by reading, not compiled, not run on hardware
//
#include <hip/hip_runtime.h>
#include <hip/hip_bf16.h>

// ---------------- problem constants (match reference) ----------------
constexpr int B_   = 4;
constexpr int S_   = 4096;
constexpr int HID_ = 2048;
constexpr int NH_  = 16;
constexpr int NKV_ = 4;
constexpr int HD_  = 128;     // HID/NH
constexpr int LAT_ = 32;      // HD/4
constexpr float EPS_ = 1e-5f;
constexpr long long MTOK = (long long)B_ * S_;   // 16384 tokens

// ---------------- types ----------------
typedef __attribute__((ext_vector_type(16))) __bf16 v16bf;
typedef __attribute__((ext_vector_type(8)))  float  v8f;

union AFrag { v16bf v; uint4 q[2]; };

__device__ inline unsigned pack2bf(float a, float b) {
    union { __bf16 h[2]; unsigned u; } x;
    x.h[0] = (__bf16)a; x.h[1] = (__bf16)b;   // lowers to v_cvt_pk_bf16_f32
    return x.u;
}

// ---------------- CDNA5 async global->LDS (ASYNCcnt) ----------------
typedef __attribute__((address_space(3))) unsigned char lds_uchar;
__device__ inline unsigned lds_off(const void* p) {
    // wave-relative LDS byte offset (async VDST operand semantics)
    return (unsigned)(unsigned long long)(const lds_uchar*)p;
}
__device__ inline void async_ld_lds_b128(void* lds_dst, const void* gsrc) {
    unsigned l = lds_off(lds_dst);
    asm volatile("global_load_async_to_lds_b128 %0, %1, off"
                 :: "v"(l), "v"(gsrc) : "memory");
}
__device__ inline void wait_async0() {
    asm volatile("s_wait_asynccnt 0x0" ::: "memory");
}

// =====================================================================
// GEMM: C[M,N] = A[M,K] @ B[K,N], fp32 in/out, bf16 WMMA compute.
// 256 threads (8 waves), block tile 128x128, LDS K-depth 64 (2 WMMA
// K-steps), ping-pong LDS + register prefetch, 16 WMMAs per barrier.
// N,K compile-time so all strides fold into immediate offsets.
// =====================================================================
#define BM 128
#define BN 128
#define BK2 64

template<int N, int K>
__global__ __launch_bounds__(256) void gemm_bf16_wmma(
    const float* __restrict__ A, const float* __restrict__ Bm,
    float* __restrict__ C, int M)
{
    __shared__ __align__(32) __bf16 ldsA[2][BM * BK2];   // [row][k]
    __shared__ __align__(32) __bf16 ldsB[2][BN * BK2];   // transposed [n][k]

    const int tid   = threadIdx.x;
    const int lane  = tid & 31;
    const int wave  = tid >> 5;        // 0..7
    const int waveM = wave >> 2;       // 0..1  (64-row slabs)
    const int waveN = wave & 3;        // 0..3  (32-col slabs)
    const int blockM = blockIdx.y * BM;
    const int blockN = blockIdx.x * BN;

    v8f acc[4][2];
    #pragma unroll
    for (int mi = 0; mi < 4; mi++)
        #pragma unroll
        for (int ni = 0; ni < 2; ni++)
            #pragma unroll
            for (int r = 0; r < 8; r++) acc[mi][ni][r] = 0.0f;

    const int lhalf = lane >> 4;       // 0: lanes 0-15, 1: lanes 16-31
    const int l16   = lane & 15;
    const int aK0   = lhalf * 16;      // A frag K 0..15 / 16..31
    const int bK0   = lhalf * 16;      // B frag K 0..15 / 16..31

    // per-thread fixed staging coordinates
    const int aRow0 = tid >> 4;              // + 16*i   (i<8)
    const int aCol  = (tid & 15) * 4;        // 4 consecutive k
    const int bN    = tid & 127;             // fixed column
    const int bK    = (tid >> 7) * 2;        // + 4*i    (i<16)

    const float* aG = A + (size_t)(blockM + aRow0) * K + aCol;   // + i*16*K + k0
    const float* bG = Bm + (size_t)bK * N + blockN + bN;          // + (k0+4i)*N

    float4 aReg[8];
    float  bReg[32];

    auto loadGlobal = [&](int k0) {
        #pragma unroll
        for (int i = 0; i < 8; i++)
            aReg[i] = *(const float4*)(aG + (size_t)i * 16 * K + k0);
        #pragma unroll
        for (int i = 0; i < 16; i++) {
            bReg[2*i]   = bG[(size_t)(k0 + 4*i) * N];
            bReg[2*i+1] = bG[(size_t)(k0 + 4*i + 1) * N];
        }
    };
    auto storeLds = [&](int buf) {
        #pragma unroll
        for (int i = 0; i < 8; i++) {
            uint2 p;
            p.x = pack2bf(aReg[i].x, aReg[i].y);
            p.y = pack2bf(aReg[i].z, aReg[i].w);
            *(uint2*)&ldsA[buf][(aRow0 + 16*i) * BK2 + aCol] = p;
        }
        #pragma unroll
        for (int i = 0; i < 16; i++)
            *(unsigned*)&ldsB[buf][bN * BK2 + bK + 4*i] = pack2bf(bReg[2*i], bReg[2*i+1]);
    };

    loadGlobal(0);
    storeLds(0);
    __syncthreads();

    int cur = 0;
    constexpr int NSTEP = K / BK2;
    for (int t = 0; t < NSTEP; ++t) {
        const bool hasNext = (t + 1) < NSTEP;
        if (hasNext) loadGlobal((t + 1) * BK2);   // loads in flight over WMMA

        #pragma unroll
        for (int kk = 0; kk < BK2; kk += 32) {
            v16bf bfrag[2];
            #pragma unroll
            for (int ni = 0; ni < 2; ni++) {
                int col = waveN * 32 + ni * 16 + l16;
                bfrag[ni] = *(const v16bf*)&ldsB[cur][col * BK2 + kk + bK0];
            }
            #pragma unroll
            for (int mi = 0; mi < 4; mi++) {
                int row = waveM * 64 + mi * 16 + l16;
                AFrag af;
                af.q[0] = *(const uint4*)&ldsA[cur][row * BK2 + kk + aK0];
                af.q[1] = *(const uint4*)&ldsA[cur][row * BK2 + kk + aK0 + 8];
                #pragma unroll
                for (int ni = 0; ni < 2; ni++) {
                    acc[mi][ni] = __builtin_amdgcn_wmma_f32_16x16x32_bf16(
                        false, af.v, false, bfrag[ni], (short)0, acc[mi][ni],
                        false, false);
                }
            }
        }
        if (hasNext) storeLds(cur ^ 1);
        __syncthreads();
        cur ^= 1;
    }

    // epilogue: one base pointer, compile-time offsets
    const int rowAdd = lhalf * 8;
    float* cptr = C + (size_t)(blockM + waveM * 64 + rowAdd) * N
                    + blockN + waveN * 32 + l16;
    #pragma unroll
    for (int mi = 0; mi < 4; mi++)
        #pragma unroll
        for (int ni = 0; ni < 2; ni++)
            #pragma unroll
            for (int r = 0; r < 8; r++)
                cptr[(size_t)(mi * 16 + r) * N + ni * 16] = acc[mi][ni][r];
}

// =====================================================================
// Kernel 2: per (token, head): optional RoPE, 128->32 latent projection,
// RMS-norm, optional elu+1. One wave per (token, head), 8 waves/block.
// Wlat staged via async global->LDS.
// =====================================================================
__global__ __launch_bounds__(256) void latent_proj(
    const float* __restrict__ src,    // [MTOK, nh, 128]
    const float* __restrict__ freqs,  // [S, 64]
    const float* __restrict__ Wlat,   // [128, 32]
    const float* __restrict__ norm_w, // [32]
    float* __restrict__ dst,          // [MTOK, nh, 32]
    int nh, int applyRope, int applyElu)
{
    __shared__ __align__(16) float sW[128 * 32];
    __shared__ float sVec[8][128];
    const int tid = threadIdx.x;
    #pragma unroll
    for (int i = 0; i < 4; i++) {
        int o = i * 256 + tid;                 // 1024 x b128
        async_ld_lds_b128(&sW[o * 4], &Wlat[o * 4]);
    }
    wait_async0();
    __syncthreads();

    const int lane = tid & 31;
    const int wave = tid >> 5;
    const long long task  = (long long)blockIdx.x * 8 + wave; // token*nh + h
    const long long token = task / nh;
    const int s = (int)(token % S_);

    const float* x = src + task * 128;
    float a0 = x[lane], a1 = x[lane + 32], a2 = x[lane + 64], a3 = x[lane + 96];
    if (applyRope) {
        float f0 = freqs[s * 64 + lane];
        float f1 = freqs[s * 64 + lane + 32];
        float c0 = cosf(f0), s0 = sinf(f0);
        float c1 = cosf(f1), s1 = sinf(f1);
        float n0 = a0 * c0 - a2 * s0;
        float n2 = a2 * c0 + a0 * s0;
        float n1 = a1 * c1 - a3 * s1;
        float n3 = a3 * c1 + a1 * s1;
        a0 = n0; a1 = n1; a2 = n2; a3 = n3;
    }
    sVec[wave][lane]      = a0;
    sVec[wave][lane + 32] = a1;
    sVec[wave][lane + 64] = a2;
    sVec[wave][lane + 96] = a3;
    __syncthreads();

    float y = 0.f;
    #pragma unroll 4
    for (int d = 0; d < 128; d++) y += sVec[wave][d] * sW[d * 32 + lane];

    float ss = y * y;
    for (int off = 16; off > 0; off >>= 1) ss += __shfl_xor(ss, off, 32);
    float r = rsqrtf(ss * (1.0f / 32.0f) + EPS_);
    y = y * r * norm_w[lane];
    if (applyElu) y = (y > 0.f) ? (y + 1.f) : __expf(y);
    dst[task * 32 + lane] = y;
}

// =====================================================================
// Kernel 3: kv[b,g] = sum_s kl^T vl (32x32), fold kvW = (kv/S) @ Wlo.
// One 1024-thread block per (g,b). Chunks staged via async LDS loads.
// =====================================================================
__global__ __launch_bounds__(1024) void kv_accum(
    const float* __restrict__ KL,   // [MTOK, NKV, 32]
    const float* __restrict__ VL,   // [MTOK, NKV, 32]
    const float* __restrict__ Wlo,  // [32, 128]
    float* __restrict__ kvW)        // [B, NKV, 32, 128]
{
    __shared__ __align__(16) float sk[64 * 32];
    __shared__ __align__(16) float sv[64 * 32];
    __shared__ float skv[32 * 32];
    const int b = blockIdx.y;
    const int g = blockIdx.x;
    const int tid = threadIdx.x;
    const int i = tid >> 5, j = tid & 31;

    const int half = tid >> 9;              // 0 -> sk, 1 -> sv
    const int o    = tid & 511;             // b128 op id (512 per array)
    const int e    = o * 4;                 // float index: sl=e>>5, jj=e&31

    float acc = 0.f;
    for (int s0 = 0; s0 < S_; s0 += 64) {
        long long tok = (long long)b * S_ + s0 + (e >> 5);
        const float* src = (half ? VL : KL) + tok * (NKV_ * 32) + g * 32 + (e & 31);
        async_ld_lds_b128(half ? &sv[e] : &sk[e], src);
        wait_async0();
        __syncthreads();
        #pragma unroll 8
        for (int cs = 0; cs < 64; cs++)
            acc += sk[cs * 32 + i] * sv[cs * 32 + j];
        __syncthreads();
    }
    skv[i * 32 + j] = acc;
    __syncthreads();

    const float invS = 1.0f / (float)S_;
    #pragma unroll
    for (int p = 0; p < 4; p++) {
        int oo = tid + p * 1024;       // 4096 outputs
        int i2 = oo >> 7;              // 0..31
        int d  = oo & 127;
        float v = 0.f;
        #pragma unroll 8
        for (int kk = 0; kk < 32; kk++) v += skv[i2 * 32 + kk] * Wlo[kk * 128 + d];
        kvW[(((long long)b * NKV_ + g) * 32 + i2) * 128 + d] = v * invS;
    }
}

// =====================================================================
// Kernel 4: out_head[d] = ql[32] @ kvW[b,g][32,128]; one wave per
// (token, head); kvW staged via async LDS loads per block.
// =====================================================================
__global__ __launch_bounds__(256) void qkv_out(
    const float* __restrict__ QL,   // [MTOK, NH, 32]
    const float* __restrict__ kvW,  // [B, NKV, 32, 128]
    float* __restrict__ attn)       // [MTOK, HID]
{
    __shared__ __align__(16) float sk[32 * 128];
    const int tid = threadIdx.x;
    const int b = blockIdx.z;
    const int h = blockIdx.y;
    const int g = h >> 2;
    const float* kw = kvW + (((long long)b * NKV_ + g) * 32) * 128;
    #pragma unroll
    for (int i = 0; i < 4; i++) {
        int o = i * 256 + tid;
        async_ld_lds_b128(&sk[o * 4], &kw[o * 4]);
    }
    wait_async0();
    __syncthreads();

    const int lane = tid & 31, wave = tid >> 5;
    const long long s   = (long long)blockIdx.x * 8 + wave;
    const long long tok = (long long)b * S_ + s;
    float qv = QL[tok * (NH_ * 32) + h * 32 + lane];
    float o0 = 0.f, o1 = 0.f, o2 = 0.f, o3 = 0.f;
    #pragma unroll 8
    for (int k = 0; k < 32; k++) {
        float qk = __shfl(qv, k, 32);
        o0 += qk * sk[k * 128 + lane];
        o1 += qk * sk[k * 128 + lane + 32];
        o2 += qk * sk[k * 128 + lane + 64];
        o3 += qk * sk[k * 128 + lane + 96];
    }
    float* o = attn + tok * HID_ + h * HD_;
    o[lane] = o0; o[lane + 32] = o1; o[lane + 64] = o2; o[lane + 96] = o3;
}

// =====================================================================
// Host-side orchestration
// =====================================================================
extern "C" void kernel_launch(void* const* d_in, const int* in_sizes, int n_in,
                              void* d_out, int out_size, void* d_ws, size_t ws_size,
                              hipStream_t stream) {
    const float* hidden = (const float*)d_in[0];
    const float* Wq     = (const float*)d_in[1];
    const float* Wk     = (const float*)d_in[2];
    const float* Wv     = (const float*)d_in[3];
    const float* Wql    = (const float*)d_in[4];
    const float* Wkl    = (const float*)d_in[5];
    const float* Wvl    = (const float*)d_in[6];
    const float* Wlo    = (const float*)d_in[7];
    const float* Wo     = (const float*)d_in[8];
    const float* norm_w = (const float*)d_in[9];
    const float* freqs  = (const float*)d_in[10];

    // workspace layout (floats); attn reuses the q slot (q dead by then)
    float* ws = (float*)d_ws;
    const long long nQ   = MTOK * (long long)(NH_ * HD_);
    const long long nKV  = MTOK * (long long)(NKV_ * HD_);
    const long long nQL  = MTOK * (long long)(NH_ * LAT_);
    const long long nKL  = MTOK * (long long)(NKV_ * LAT_);
    float* qbuf = ws;
    float* kbuf = qbuf + nQ;
    float* vbuf = kbuf + nKV;
    float* ql   = vbuf + nKV;
    float* kl   = ql + nQL;
    float* vl   = kl + nKL;
    float* kvW  = vl + nKL;                 // B*NKV*32*128
    float* attn = qbuf;                     // reuse

    const int M = (int)MTOK;

    // 1) Q/K/V projections (WMMA bf16)
    gemm_bf16_wmma<2048, 2048><<<dim3((NH_ * HD_) / BN, M / BM), 256, 0, stream>>>(
        hidden, Wq, qbuf, M);
    gemm_bf16_wmma<512, 2048><<<dim3((NKV_ * HD_) / BN, M / BM), 256, 0, stream>>>(
        hidden, Wk, kbuf, M);
    gemm_bf16_wmma<512, 2048><<<dim3((NKV_ * HD_) / BN, M / BM), 256, 0, stream>>>(
        hidden, Wv, vbuf, M);

    // 2) RoPE + latent projection + RMS-norm (+elu+1 for q,k)
    latent_proj<<<(unsigned)(MTOK * NH_ / 8), 256, 0, stream>>>(
        qbuf, freqs, Wql, norm_w, ql, NH_, 1, 1);
    latent_proj<<<(unsigned)(MTOK * NKV_ / 8), 256, 0, stream>>>(
        kbuf, freqs, Wkl, norm_w, kl, NKV_, 1, 1);
    latent_proj<<<(unsigned)(MTOK * NKV_ / 8), 256, 0, stream>>>(
        vbuf, freqs, Wvl, norm_w, vl, NKV_, 0, 0);

    // 3) kv state + fold Wlo & 1/S
    kv_accum<<<dim3(NKV_, B_), 1024, 0, stream>>>(kl, vl, Wlo, kvW);

    // 4) per-token head outputs into attn (= q slot)
    qkv_out<<<dim3(S_ / 8, NH_, B_), 256, 0, stream>>>(ql, kvW, attn);

    // 5) output projection (WMMA bf16) -> d_out
    gemm_bf16_wmma<2048, 2048><<<dim3(HID_ / BN, M / BM), 256, 0, stream>>>(
        attn, Wo, (float*)d_out, M);
}
